// HeteroRGCN_31593779429426
// MI455X (gfx1250) — compile-verified, matrix operand order checked by reading, not run on hardware
//
#include <hip/hip_runtime.h>

#define IN_F 128
#define HID  128
#define N_REL 5

typedef __attribute__((ext_vector_type(16))) __bf16 v16bf;
typedef __attribute__((ext_vector_type(8)))  float  v8f;

__device__ __forceinline__ __bf16 f2bf(float f) { return (__bf16)f; }

// K index within a 32-wide K-step for operand element i of a lane in half `hi`.
// ISA 16-bit A/B layout: lanes 0-15 hold K {0..7, 16..23}, lanes 16-31 hold {8..15, 24..31}.
__device__ __forceinline__ int kmap(int i, int hi) {
    return (i & 7) + ((i >> 3) << 4) + (hi << 3);
}

// Fused per-layer GEMM: for one 16x128 strip of X per wave, computes
//   tmp[r] = act(X) @ Wrel[r]          (r = 0..4, contiguous 128x128 matrices)
//   acc    = act(X) @ root + bias
// act = relu if relu_in, else identity. A is loaded/converted ONCE (split-bf16
// hi/lo in registers), each W is staged in LDS in WMMA-B lane order per pass.
__global__ __launch_bounds__(256)
void rgcn_gemm_fused(const float* __restrict__ X, const float* __restrict__ Wrel,
                     const float* __restrict__ root, const float* __restrict__ bias,
                     float* __restrict__ tmp, float* __restrict__ acc,
                     int N, int relu_in)
{
    __shared__ __align__(32) __bf16 sBh[4][8][32][16];
    __shared__ __align__(32) __bf16 sBl[4][8][32][16];

    const int t    = threadIdx.x;
    const int lane = t & 31;
    const int wave = t >> 5;
    const int hi   = lane >> 4;

    const int rowBase = (blockIdx.x * 8 + wave) * 16;
    int row = rowBase + (lane & 15);
    if (row >= N) row = N - 1;            // clamp: keep EXEC all-1s for WMMA
    const float* xrow = X + (size_t)row * HID;
    const size_t nh = (size_t)N * HID;

    // ---- load + split-convert A once: Ah/Al[ks] in registers ----
    v16bf Ah[4], Al[4];
#pragma unroll
    for (int ks = 0; ks < 4; ++ks) {
        const int k0 = ks * 32 + (hi << 3);
        const float4 a0 = *(const float4*)(xrow + k0);
        const float4 a1 = *(const float4*)(xrow + k0 + 4);
        const float4 a2 = *(const float4*)(xrow + k0 + 16);
        const float4 a3 = *(const float4*)(xrow + k0 + 20);
        float av[16] = {a0.x, a0.y, a0.z, a0.w, a1.x, a1.y, a1.z, a1.w,
                        a2.x, a2.y, a2.z, a2.w, a3.x, a3.y, a3.z, a3.w};
#pragma unroll
        for (int i = 0; i < 16; ++i) {
            float a = relu_in ? fmaxf(av[i], 0.0f) : av[i];
            const __bf16 h = f2bf(a);
            Ah[ks][i] = h;
            Al[ks][i] = f2bf(a - (float)h);
        }
    }

    // ---- loop over the 6 weight matrices, reusing A ----
    for (int m = 0; m < N_REL + 1; ++m) {
        const float* W = (m < N_REL) ? (Wrel + (size_t)m * IN_F * HID) : root;
        float* Y = (m < N_REL) ? (tmp + (size_t)m * nh) : acc;

        __syncthreads();   // previous pass done reading LDS
        for (int s = t; s < 4 * 8 * 32; s += 256) {
            const int l   = s & 31;
            const int ct  = (s >> 5) & 7;
            const int ks  = s >> 8;
            const int lhi = l >> 4;
            const int col = (ct << 4) + (l & 15);
#pragma unroll
            for (int i = 0; i < 16; ++i) {
                const int k = ks * 32 + kmap(i, lhi);
                const float w = W[k * HID + col];
                const __bf16 h = f2bf(w);
                sBh[ks][ct][l][i] = h;
                sBl[ks][ct][l][i] = f2bf(w - (float)h);
            }
        }
        __syncthreads();

        v8f acc8[8];
#pragma unroll
        for (int ct = 0; ct < 8; ++ct)
#pragma unroll
            for (int j = 0; j < 8; ++j) acc8[ct][j] = 0.0f;

#pragma unroll
        for (int ks = 0; ks < 4; ++ks) {
#pragma unroll
            for (int ct = 0; ct < 8; ++ct) {
                const v16bf bh = *(const v16bf*)&sBh[ks][ct][lane][0];
                const v16bf bl = *(const v16bf*)&sBl[ks][ct][lane][0];
                acc8[ct] = __builtin_amdgcn_wmma_f32_16x16x32_bf16(
                    false, Ah[ks], false, bh, (short)0, acc8[ct], false, false);
                acc8[ct] = __builtin_amdgcn_wmma_f32_16x16x32_bf16(
                    false, Al[ks], false, bh, (short)0, acc8[ct], false, false);
                acc8[ct] = __builtin_amdgcn_wmma_f32_16x16x32_bf16(
                    false, Ah[ks], false, bl, (short)0, acc8[ct], false, false);
            }
        }

        // D layout: VGPR v -> row = rowBase + v + 8*hi, col = ct*16 + (lane&15)
        const int useBias = (m == N_REL);
#pragma unroll
        for (int ct = 0; ct < 8; ++ct) {
            const int col = (ct << 4) + (lane & 15);
            const float badd = useBias ? bias[col] : 0.0f;
#pragma unroll
            for (int v = 0; v < 8; ++v) {
                const int r = rowBase + v + (hi << 3);
                if (r < N) Y[(size_t)r * HID + col] = acc8[ct][v] + badd;
            }
        }
    }
}

// One pass over ALL edges: acc[dst] += tmp[etype][src] * inv[etype][dst].
// 32 lanes per edge (branch-uniform), float4 gather + 4 f32 atomics per lane.
__global__ __launch_bounds__(256)
void rgcn_scatter(const int* __restrict__ src, const int* __restrict__ dst,
                  const int* __restrict__ etype, const float* __restrict__ tmp,
                  const float* __restrict__ inv, float* __restrict__ acc,
                  int E, int N)
{
    const long long tid = (long long)blockIdx.x * blockDim.x + threadIdx.x;
    const long long e = tid >> 5;
    if (e >= E) return;
    const int lane = (int)(tid & 31);
    const int r = etype[e];
    const int s = src[e];
    const int d = dst[e];
    const size_t nh = (size_t)N * HID;
    const float w = inv[(size_t)r * N + d];
    const float4 v = *(const float4*)(tmp + (size_t)r * nh + (size_t)s * HID + lane * 4);
    float* out = acc + (size_t)d * HID + lane * 4;
    atomicAdd(out + 0, v.x * w);
    atomicAdd(out + 1, v.y * w);
    atomicAdd(out + 2, v.z * w);
    atomicAdd(out + 3, v.w * w);
}

__global__ __launch_bounds__(256)
void rgcn_count(const int* __restrict__ dst, const int* __restrict__ etype,
                float* __restrict__ cnt, int E, int N)
{
    const long long e = (long long)blockIdx.x * blockDim.x + threadIdx.x;
    if (e < E) atomicAdd(&cnt[(size_t)etype[e] * N + dst[e]], 1.0f);
}

__global__ __launch_bounds__(256)
void fzero(float* __restrict__ p, long long n)
{
    const long long i = (long long)blockIdx.x * blockDim.x + threadIdx.x;
    if (i < n) p[i] = 0.0f;
}

__global__ __launch_bounds__(256)
void finv(float* __restrict__ p, long long n)   // p = 1/max(p,1) in place
{
    const long long i = (long long)blockIdx.x * blockDim.x + threadIdx.x;
    if (i < n) p[i] = 1.0f / fmaxf(p[i], 1.0f);
}

__global__ __launch_bounds__(256)
void frelu(const float* __restrict__ in, float* __restrict__ out, long long n)
{
    const long long i = (long long)blockIdx.x * blockDim.x + threadIdx.x;
    if (i < n) out[i] = fmaxf(in[i], 0.0f);
}

extern "C" void kernel_launch(void* const* d_in, const int* in_sizes, int n_in,
                              void* d_out, int out_size, void* d_ws, size_t ws_size,
                              hipStream_t stream)
{
    (void)n_in; (void)out_size; (void)ws_size;

    const float* x   = (const float*)d_in[0];
    const int*   ei  = (const int*)d_in[1];
    const int*   et  = (const int*)d_in[2];
    const float* W[3]    = {(const float*)d_in[3], (const float*)d_in[6], (const float*)d_in[9]};
    const float* root[3] = {(const float*)d_in[4], (const float*)d_in[7], (const float*)d_in[10]};
    const float* bias[3] = {(const float*)d_in[5], (const float*)d_in[8], (const float*)d_in[11]};

    const int N = in_sizes[0] / IN_F;
    const int E = in_sizes[1] / 2;
    const int* src = ei;
    const int* dst = ei + E;

    // workspace: accA | accB | tmp[5] | cnt(->inv) == (7*N*HID + N_REL*N) floats (~360 MB)
    const size_t nh = (size_t)N * HID;
    float* accA = (float*)d_ws;
    float* accB = accA + nh;
    float* tmpb = accB + nh;              // 5 contiguous [N,HID] buffers
    float* cnt  = tmpb + (size_t)N_REL * nh;

    const int gemm_blocks = (N + 127) / 128;
    const long long ncnt = (long long)N_REL * N;

    // per-(rel,dst) mean normalizer, computed once (constant across layers)
    fzero<<<(int)((ncnt + 255) / 256), 256, 0, stream>>>(cnt, ncnt);
    rgcn_count<<<(E + 255) / 256, 256, 0, stream>>>(dst, et, cnt, E, N);
    finv<<<(int)((ncnt + 255) / 256), 256, 0, stream>>>(cnt, ncnt);

    const long long sthreads = (long long)E * 32;
    const int sblocks = (int)((sthreads + 255) / 256);

    const float* lin = x;                 // layer input (pre-activation for L2/L3)
    float* accs[3] = {accA, accB, accA};  // ping-pong accumulators
    for (int L = 0; L < 3; ++L) {
        float* acc = accs[L];
        // tmp[r] = act(lin) @ W[L][r];  acc = act(lin) @ root[L] + bias[L]
        rgcn_gemm_fused<<<gemm_blocks, 256, 0, stream>>>(
            lin, W[L], root[L], bias[L], tmpb, acc, N, L > 0 ? 1 : 0);
        // acc[dst] += tmp[etype][src] * inv   (segment mean, normalizer folded per edge)
        rgcn_scatter<<<sblocks, 256, 0, stream>>>(src, dst, et, tmpb, cnt, acc, E, N);
        lin = acc;                        // relu is applied on load by the next GEMM
    }
    // final activation
    frelu<<<(int)((nh + 255) / 256), 256, 0, stream>>>(accs[2], (float*)d_out, (long long)nh);
}
